// H2OAttention_51625506898367
// MI455X (gfx1250) — matrix operand order repaired and checked by reading, not compiled, not run on hardware
//
#include <hip/hip_runtime.h>
#include <hip/hip_fp16.h>

typedef __attribute__((ext_vector_type(16))) _Float16 v16h;
typedef __attribute__((ext_vector_type(8)))  _Float16 v8h;
typedef __attribute__((ext_vector_type(4)))  _Float16 v4h;
typedef __attribute__((ext_vector_type(8)))  float    v8f;
typedef __attribute__((ext_vector_type(4)))  unsigned u32x4;
typedef __attribute__((ext_vector_type(8)))  unsigned u32x8;

#define S_TOK 2048
#define HDIM  1024
#define NH    16
#define HD    64
#define WIN   1024

// ---------- CDNA5 async copy (ASYNCcnt path) ----------
__device__ inline void async_copy_b128(unsigned ldsOff, const _Float16* g) {
  asm volatile("global_load_async_to_lds_b128 %0, %1, off"
               :: "v"(ldsOff), "v"(g) : "memory");
}
__device__ inline void wait_async0() {
  asm volatile("s_wait_asynccnt 0" ::: "memory");
}

// ---------- CDNA5 Tensor Data Mover: 64-row x 32-elem f16 tile, row stride
// 2048 elems, from global to LDS (D# per ISA 8.3/8.4; 2-D -> groups 0,1 only)
__device__ inline void tdm_load_tile64x32(unsigned ldsOff, const _Float16* g) {
  const unsigned long long ga = (unsigned long long)(size_t)g;
  union { u32x4 v; unsigned w[4]; } g0;
  union { u32x8 v; unsigned w[8]; } g1;
  g0.w[0] = 1u;                                   // count=1 (valid descriptor)
  g0.w[1] = ldsOff;                               // lds_addr (bytes)
  g0.w[2] = (unsigned)ga;                         // global_addr[31:0]
  g0.w[3] = (unsigned)((ga >> 32) & 0x1ffffffu)   // global_addr[56:32]
          | (2u << 30);                           // type = 2 ("image")
  g1.w[0] = 1u << 16;                             // data_size = 1 -> 2 bytes
  g1.w[1] = 0x08000000u;                          // tensor_dim0 = 2048 (lo16 in [63:48])
  g1.w[2] = 0u;                                   // dim0 hi16 = 0, dim1 lo16 = 0
  g1.w[3] = (32u << 16) | 0x0010u;                // tile_dim0=32, tensor_dim1=1<<20
  g1.w[4] = 64u;                                  // tile_dim1=64, tile_dim2=0
  g1.w[5] = 2048u;                                // tensor_dim0_stride lo32
  g1.w[6] = 0u;                                   // stride hi16, dim1_stride lo16
  g1.w[7] = 0u;
  asm volatile("tensor_load_to_lds %0, %1" :: "s"(g0.v), "s"(g1.v) : "memory");
}

// ---------- WMMA fragment helpers (ISA 7.12.2 layouts, wave32) ----------
__device__ inline v16h load_A16x32(const _Float16* __restrict__ base, int ld) {
  const int lane = threadIdx.x & 31;
  const int m  = lane & 15;
  const int k0 = (lane >> 4) << 3;                 // 0 or 8
  const _Float16* p = base + m * ld + k0;
  v8h lo = *(const v8h*)(p);
  v8h hi = *(const v8h*)(p + 16);
  v16h a;
#pragma unroll
  for (int i = 0; i < 8; ++i) { a[i] = lo[i]; a[i + 8] = hi[i]; }
  return a;
}

__device__ inline v16h load_B32x16(const _Float16* base, int ld) {
  const int lane = threadIdx.x & 31;
  const int n  = lane & 15;
  const int kh = (lane >> 4) << 3;                 // 0 or 8
  const _Float16* p = base + n * ld + kh;
  v8h lo = *(const v8h*)(p);
  v8h hi = *(const v8h*)(p + 16);
  v16h b;
#pragma unroll
  for (int i = 0; i < 8; ++i) { b[i] = lo[i]; b[i + 8] = hi[i]; }
  return b;
}

__device__ inline v8f wmma16(v16h a, v16h b, v8f c) {
  return __builtin_amdgcn_wmma_f32_16x16x32_f16(false, a, false, b,
                                                (short)0, c, false, false);
}

// ---------- f32 -> f16 cast ----------
__global__ void cvt_f32_f16(const float* __restrict__ in,
                            _Float16* __restrict__ out, int n4) {
  int i = blockIdx.x * blockDim.x + threadIdx.x;
  if (i < n4) {
    float4 x = ((const float4*)in)[i];
    v4h y;
    y[0] = (_Float16)x.x; y[1] = (_Float16)x.y;
    y[2] = (_Float16)x.z; y[3] = (_Float16)x.w;
    ((v4h*)out)[i] = y;
  }
}

// ---------- GEMM: Y[s][o] = sum_h X[s][h] * W[o][h] ------------------------
// 4-wave block, 64x64 tile; B panel staged via async-to-LDS, double-buffered.
template <int MODE>
__global__ void gemm_xwT(const _Float16* __restrict__ X,
                         const _Float16* __restrict__ W,
                         _Float16* __restrict__ outH,
                         float* __restrict__ outF) {
  __shared__ __align__(16) _Float16 Bbuf[2][64 * 32];
  const int wave = threadIdx.x >> 5;
  const int sblk = blockIdx.x & 31;
  const int oblk = blockIdx.x >> 5;
  const int srow = sblk * 64 + wave * 16;
  const int ocol = oblk * 64;
  const int lane = threadIdx.x & 31;
  const int n  = lane & 15;
  const int hi = lane >> 4;

  const unsigned ldsB0 = (unsigned)(size_t)(&Bbuf[0][0]);

  auto stage = [&](int buf, int k0) {
    const int t = threadIdx.x;
#pragma unroll
    for (int j = 0; j < 2; ++j) {
      const int seg = t * 2 + j;
      const int r   = seg >> 2;
      const int off = (seg & 3) << 3;
      async_copy_b128(ldsB0 + (unsigned)((buf * 64 * 32 + r * 32 + off) * 2),
                      W + (size_t)(ocol + r) * HDIM + k0 + off);
    }
  };

  stage(0, 0);
  wait_async0();
  __syncthreads();

  v8f acc[4] = {};
  for (int kc = 0; kc < HDIM / 32; ++kc) {
    const int cur = kc & 1;
    if (kc + 1 < HDIM / 32) stage(cur ^ 1, (kc + 1) * 32);

    v16h a = load_A16x32(X + (size_t)srow * HDIM + kc * 32, HDIM);
#pragma unroll
    for (int t = 0; t < 4; ++t) {
      v16h b = load_B32x16(&Bbuf[cur][t * 16 * 32], 32);
      acc[t] = wmma16(a, b, acc[t]);
    }
    wait_async0();
    __syncthreads();
  }

  if (MODE == 0) {
#pragma unroll
    for (int t = 0; t < 4; ++t)
#pragma unroll
      for (int v = 0; v < 8; ++v) {
        const int row = srow + v + 8 * hi;
        outH[(size_t)row * HDIM + ocol + t * 16 + n] = (_Float16)acc[t][v];
      }
  } else if (MODE == 1) {
#pragma unroll
    for (int t = 0; t < 4; ++t) {
      v8h pk;
#pragma unroll
      for (int v = 0; v < 8; ++v) pk[v] = (_Float16)acc[t][v];
      *(v8h*)(outH + (size_t)(ocol + t * 16 + n) * S_TOK + srow + 8 * hi) = pk;
    }
  } else {
#pragma unroll
    for (int t = 0; t < 4; ++t)
#pragma unroll
      for (int v = 0; v < 8; ++v) {
        const int row = srow + v + 8 * hi;
        outF[(size_t)row * HDIM + ocol + t * 16 + n] = acc[t][v];
      }
  }
}

// ---------- attention step (templated so full steps carry no mask VALU) ----
// Scores arrive pre-scaled by 0.125*log2(e) (folded into Q), softmax in
// exp2 domain -> bare v_exp_f32. WAITN = s_wait_tensorcnt argument before
// consuming the TDM-staged V^T tile.
template <bool MASKED, int WAITN>
__device__ __forceinline__ void attn_step(
    int j0, int r0, int lane,
    v16h aq0, v16h aq1,
    const _Float16* __restrict__ kb,       // K + j0*HDIM + head*HD
    const _Float16* vcur,                  // LDS: 64x32 f16 V^T tile
    float* sbufw, float* abufw,
    float& mrow, float& lrow, v8f* o) {
  const int n  = lane & 15;
  const int hi = lane >> 4;
  const int mm = n;
  const int koff = hi << 3;

  v8f s0 = {}, s1 = {};
  s0 = wmma16(aq0, load_B32x16(kb, HDIM), s0);
  s0 = wmma16(aq1, load_B32x16(kb + 32, HDIM), s0);
  s1 = wmma16(aq0, load_B32x16(kb + 16 * HDIM, HDIM), s1);
  s1 = wmma16(aq1, load_B32x16(kb + 16 * HDIM + 32, HDIM), s1);

#pragma unroll
  for (int v = 0; v < 8; ++v) {
    const int m = v + 8 * hi;
    float x0 = s0[v];
    float x1 = s1[v];
    if (MASKED) {
      const int r = r0 + m;
      if (j0 + n > r)      x0 = -3.0e38f;
      if (j0 + 16 + n > r) x1 = -3.0e38f;
    }
    sbufw[m * 32 + n]      = x0;
    sbufw[m * 32 + 16 + n] = x1;
  }

  const float* sp = sbufw + mm * 32 + koff;
  float4 q0 = *(const float4*)(sp);
  float4 q1 = *(const float4*)(sp + 4);
  float4 q2 = *(const float4*)(sp + 16);
  float4 q3 = *(const float4*)(sp + 20);
  float x[16] = {q0.x, q0.y, q0.z, q0.w, q1.x, q1.y, q1.z, q1.w,
                 q2.x, q2.y, q2.z, q2.w, q3.x, q3.y, q3.z, q3.w};

  float tmax = x[0];
#pragma unroll
  for (int i = 1; i < 16; ++i) tmax = fmaxf(tmax, x[i]);
  tmax = fmaxf(tmax, __shfl_xor(tmax, 16, 32));
  const float mnew = fmaxf(mrow, tmax);

  float p[16], lsum = 0.0f;
#pragma unroll
  for (int i = 0; i < 16; ++i) { p[i] = exp2f(x[i] - mnew); lsum += p[i]; }
  lsum += __shfl_xor(lsum, 16, 32);
  const float alpha = exp2f(mrow - mnew);
  lrow = lrow * alpha + lsum;
  mrow = mnew;

  v16h ap;
#pragma unroll
  for (int i = 0; i < 16; ++i) ap[i] = (_Float16)p[i];

  abufw[mm] = alpha;
  const float4 a0 = *(const float4*)(abufw + hi * 8);
  const float4 a1 = *(const float4*)(abufw + hi * 8 + 4);
  const float av[8] = {a0.x, a0.y, a0.z, a0.w, a1.x, a1.y, a1.z, a1.w};
#pragma unroll
  for (int dt = 0; dt < 4; ++dt)
#pragma unroll
    for (int v = 0; v < 8; ++v) o[dt][v] *= av[v];

  __builtin_amdgcn_s_wait_tensorcnt(WAITN);    // V^T tile resident in LDS
#pragma unroll
  for (int dt = 0; dt < 4; ++dt) {
    v16h bv = load_B32x16(vcur + dt * 16 * 32, 32);
    o[dt] = wmma16(ap, bv, o[dt]);
  }
}

// ---------- Flash attention per (head, 16-query tile) -----------------------
// Reference semantics: query q attends keys 0..(q mod 1024).
__global__ void attn_kernel(const _Float16* __restrict__ Q,
                            const _Float16* __restrict__ K,
                            const _Float16* __restrict__ VT,   // [H][S] f16
                            _Float16* __restrict__ O) {
  __shared__ __align__(16) float    sbuf[4][16 * 32];
  __shared__ __align__(16) float    abuf[4][16];
  __shared__ __align__(16) _Float16 vbuf[4][2][64 * 32];   // TDM double buffer
  const int wave = threadIdx.x >> 5;
  const int tile = blockIdx.x * (blockDim.x >> 5) + wave;
  const int head = tile >> 7;
  const int srow = (tile & 127) << 4;
  const int r0   = srow & (WIN - 1);
  const int lane = threadIdx.x & 31;
  const int n  = lane & 15;
  const int hi = lane >> 4;

  v16h aq0 = load_A16x32(Q + (size_t)srow * HDIM + head * HD, HDIM);
  v16h aq1 = load_A16x32(Q + (size_t)srow * HDIM + head * HD + 32, HDIM);
  const _Float16 qs = (_Float16)(0.125f * 1.44269504f);  // scale * log2(e)
#pragma unroll
  for (int i = 0; i < 16; ++i) { aq0[i] = aq0[i] * qs; aq1[i] = aq1[i] * qs; }

  float* sbufw = &sbuf[wave][0];
  float* abufw = &abuf[wave][0];
  const _Float16* vtbase = VT + (size_t)head * HD * S_TOK;
  const unsigned vlds = (unsigned)(size_t)(&vbuf[wave][0][0]);
  const unsigned VB   = 64 * 32 * 2;                     // bytes per buffer

  v8f o[4] = {};
  float mrow = -3.0e38f, lrow = 0.0f;

  tdm_load_tile64x32(vlds, vtbase);                      // keys 0..31 of V^T
  int buf = 0;
  const int nfull = r0 >> 5;                             // mask-free steps
  for (int s = 0; s < nfull; ++s) {
    const int j0 = s * 32;
    tdm_load_tile64x32(vlds + (unsigned)((buf ^ 1) * VB), vtbase + j0 + 32);
    attn_step<false, 1>(j0, r0, lane, aq0, aq1,
                        K + (size_t)j0 * HDIM + head * HD,
                        &vbuf[wave][buf][0], sbufw, abufw, mrow, lrow, o);
    buf ^= 1;
  }
  {
    const int j0 = nfull * 32;                           // single masked tail
    attn_step<true, 0>(j0, r0, lane, aq0, aq1,
                       K + (size_t)j0 * HDIM + head * HD,
                       &vbuf[wave][buf][0], sbufw, abufw, mrow, lrow, o);
  }

  abufw[n] = lrow;
  const float4 l0 = *(const float4*)(abufw + hi * 8);
  const float4 l1 = *(const float4*)(abufw + hi * 8 + 4);
  const float lv[8] = {l0.x, l0.y, l0.z, l0.w, l1.x, l1.y, l1.z, l1.w};
#pragma unroll
  for (int dt = 0; dt < 4; ++dt)
#pragma unroll
    for (int v = 0; v < 8; ++v) {
      const int row = srow + v + 8 * hi;
      O[(size_t)row * HDIM + head * HD + dt * 16 + n] =
          (_Float16)(o[dt][v] / lv[v]);
    }
}

// ---------- host launch ----------
extern "C" void kernel_launch(void* const* d_in, const int* in_sizes, int n_in,
                              void* d_out, int out_size, void* d_ws, size_t ws_size,
                              hipStream_t stream) {
  (void)in_sizes; (void)n_in; (void)out_size; (void)ws_size;
  const float* X  = (const float*)d_in[0];
  const float* Wq = (const float*)d_in[1];
  const float* Wk = (const float*)d_in[2];
  const float* Wv = (const float*)d_in[3];
  const float* Wo = (const float*)d_in[4];

  char* ws = (char*)d_ws;
  _Float16* Xh  = (_Float16*)(ws + (0ull  << 20));
  _Float16* Wqh = (_Float16*)(ws + (4ull  << 20));
  _Float16* Wkh = (_Float16*)(ws + (6ull  << 20));
  _Float16* Wvh = (_Float16*)(ws + (8ull  << 20));
  _Float16* Woh = (_Float16*)(ws + (10ull << 20));
  _Float16* Qh  = (_Float16*)(ws + (12ull << 20));
  _Float16* Kh  = (_Float16*)(ws + (16ull << 20));
  _Float16* VT  = (_Float16*)(ws + (20ull << 20));
  _Float16* Ah  = (_Float16*)(ws + (24ull << 20));

  const int nX = S_TOK * HDIM / 4, nW = HDIM * HDIM / 4;
  cvt_f32_f16<<<(nX + 255) / 256, 256, 0, stream>>>(X,  Xh,  nX);
  cvt_f32_f16<<<(nW + 255) / 256, 256, 0, stream>>>(Wq, Wqh, nW);
  cvt_f32_f16<<<(nW + 255) / 256, 256, 0, stream>>>(Wk, Wkh, nW);
  cvt_f32_f16<<<(nW + 255) / 256, 256, 0, stream>>>(Wv, Wvh, nW);
  cvt_f32_f16<<<(nW + 255) / 256, 256, 0, stream>>>(Wo, Woh, nW);

  gemm_xwT<0><<<512, 128, 0, stream>>>(Xh, Wqh, Qh, nullptr);
  gemm_xwT<0><<<512, 128, 0, stream>>>(Xh, Wkh, Kh, nullptr);
  gemm_xwT<1><<<512, 128, 0, stream>>>(Xh, Wvh, VT, nullptr);

  attn_kernel<<<512, 128, 0, stream>>>(Qh, Kh, VT, Ah);

  gemm_xwT<2><<<512, 128, 0, stream>>>(Ah, Woh, nullptr, (float*)d_out);
}
